// TimeMixing_70334384439529
// MI455X (gfx1250) — compile-verified, matrix operand order checked by reading, not run on hardware
//
#include <hip/hip_runtime.h>
#include <hip/hip_bf16.h>
#include <stdint.h>

// ---------------------------------------------------------------------------
// RWKV-ish TimeMixing for MI455X (gfx1250, wave32)
//
// Pipeline (all on `stream`, graph-capture safe):
//   0a prep_scalars : sigmoid(gates), exp(time_first), w=exp(decay), w^128
//   0b prep_pack    : key_w/value_w f32 -> bf16, pre-swizzled into the exact
//                     per-lane V_WMMA_F32_16X16X32_BF16 B-fragment order
//   1  fused_shift_ln_kv : token shift + LayerNorm + K/V projection (bf16 WMMA,
//                     f32 accum) + wkv = exp(-exp(tf)*k)*v   -> ws (b,s,n)
//   2  scan_pass1/2/3 : chunked parallel linear recurrence (32 chunks x 128),
//                     pass2 also emits last_states
//   3  out_proj     : states @ output_w^T with V_WMMA_F32_16X16X4_F32 (f32),
//                     LDS-staged coalesced stores
// ---------------------------------------------------------------------------

#define B_      8
#define S_      4096
#define D_      1024
#define N_      64
#define SHIFT_  2048          // int(S * 0.5)
#define LN_EPS  1e-5f
#define NCHUNK  32
#define CHUNK   128           // S_/NCHUNK
#define PADB    8             // LDS row pad (halfwords) -> conflict-free ds_read_b128
#define ROWB    (D_ + PADB)   // 1032 halfwords per staged activation row

typedef __attribute__((ext_vector_type(16))) __bf16 v16bf;
typedef __attribute__((ext_vector_type(8)))  float  v8f;
typedef __attribute__((ext_vector_type(2)))  float  v2f;

static __device__ __forceinline__ unsigned short f32_to_bf16_rne(float f) {
  unsigned int u = __float_as_uint(f);
  u += 0x7fffu + ((u >> 16) & 1u);       // round-to-nearest-even
  return (unsigned short)(u >> 16);
}

// ---------------------------------------------------------------------------
// 0a: per-channel scalars
// ---------------------------------------------------------------------------
__global__ __launch_bounds__(1024) void prep_scalars(
    const float* __restrict__ td, const float* __restrict__ tf_in,
    const float* __restrict__ gates,
    float* __restrict__ g_sig, float* __restrict__ tf,
    float* __restrict__ wn, float* __restrict__ wc) {
  int i = threadIdx.x;
  if (i < D_) g_sig[i] = 1.0f / (1.0f + __expf(-gates[i]));
  if (i < N_) {
    float w = __expf(td[i]);
    wn[i] = w;
    tf[i] = __expf(tf_in[i]);
    float t = w;                          // w^(2^7) = w^CHUNK
    #pragma unroll
    for (int k = 0; k < 7; ++k) t *= t;
    wc[i] = t;
  }
}

// ---------------------------------------------------------------------------
// 0b: pack key_w/value_w into bf16 B-fragment order for 16x16x32 WMMA.
// B (32x16, K x N) 16-bit lane layout (CDNA5 ISA, mirrored from A):
//   lanes 0-15 : N=lane,    K in {0..7, 16..23}
//   lanes 16-31: N=lane-16, K in {8..15, 24..31}
// per-lane 16 bf16: elems 0..7 -> first K-octet, elems 8..15 -> second.
// flat index = (((mat*4 + ntile)*32 + kstep)*32 + lane)*16 + e
// ---------------------------------------------------------------------------
__global__ __launch_bounds__(256) void prep_pack(
    const float* __restrict__ key_w, const float* __restrict__ value_w,
    unsigned short* __restrict__ packed) {
  int idx = blockIdx.x * 256 + threadIdx.x;        // 2*4*32*32*16 = 131072
  if (idx >= 131072) return;
  int e    = idx & 15;
  int lane = (idx >> 4) & 31;
  int ks   = (idx >> 9) & 31;
  int tn   = (idx >> 14) & 3;
  int m    = idx >> 16;
  int n    = tn * 16 + (lane & 15);
  int grp  = lane >> 4;
  int k    = ks * 32 + ((e & 8) ? 16 : 0) + grp * 8 + (e & 7);
  const float* src = m ? value_w : key_w;
  packed[idx] = f32_to_bf16_rne(src[n * D_ + k]);
}

// ---------------------------------------------------------------------------
// 1: fused token-shift + LayerNorm + K/V projection + wkv
// 256 threads (8 waves). 16-token row tile. waves 0-3: key n-tiles 0-3,
// waves 4-7: value n-tiles 0-3. 32 k-steps of v_wmma_f32_16x16x32_bf16.
// ---------------------------------------------------------------------------
__global__ __launch_bounds__(256) void fused_shift_ln_kv(
    const float* __restrict__ x,
    const float* __restrict__ lnw, const float* __restrict__ lnb,
    const float* __restrict__ g_sig,
    const unsigned short* __restrict__ packed,
    const float* __restrict__ tf,
    float* __restrict__ wkv) {
  __shared__ __align__(16) unsigned short xnb[16 * ROWB];   // bf16 activations
  __shared__ float red0[16][16];
  __shared__ float red1[16][16];
  __shared__ float2 rowstat[16];
  __shared__ float kvbuf[2][16][64];                        // key / value tiles

  const int tid = threadIdx.x;
  const int r0  = blockIdx.x * 16;
  const int b   = r0 >> 12;                 // S_ = 4096
  const int s0  = r0 & (S_ - 1);

  // ---- phase 1: shift + stats (16 threads per row, 16 float4 each) ----
  const int row = tid >> 4;
  const int sub = tid & 15;
  const int t   = s0 + row;
  const int t2  = (t + SHIFT_) & (S_ - 1);  // circular roll by +shift
  const float* xr  = x + (size_t)(b * S_ + t)  * D_;
  const float* xr2 = x + (size_t)(b * S_ + t2) * D_;

  float4 xs_loc[16];
  float sum = 0.f, sq = 0.f;
  #pragma unroll
  for (int j = 0; j < 16; ++j) {
    int c = (sub + j * 16) * 4;
    float4 a  = *(const float4*)(xr  + c);
    float4 a2 = *(const float4*)(xr2 + c);
    float4 g4 = *(const float4*)(g_sig + c);
    float4 xs;
    xs.x = a2.x * g4.x + a.x * (1.f - g4.x);
    xs.y = a2.y * g4.y + a.y * (1.f - g4.y);
    xs.z = a2.z * g4.z + a.z * (1.f - g4.z);
    xs.w = a2.w * g4.w + a.w * (1.f - g4.w);
    sum += xs.x + xs.y + xs.z + xs.w;
    sq  += xs.x * xs.x + xs.y * xs.y + xs.z * xs.z + xs.w * xs.w;
    xs_loc[j] = xs;
  }
  red0[row][sub] = sum;
  red1[row][sub] = sq;
  __syncthreads();
  if (tid < 16) {
    float s = 0.f, q = 0.f;
    #pragma unroll
    for (int i = 0; i < 16; ++i) { s += red0[tid][i]; q += red1[tid][i]; }
    float mu  = s * (1.0f / D_);
    float var = q * (1.0f / D_) - mu * mu;   // biased variance
    rowstat[tid] = make_float2(mu, rsqrtf(var + LN_EPS));
  }
  __syncthreads();

  // ---- phase 2: normalize -> bf16 into LDS ----
  {
    float mu = rowstat[row].x, rs = rowstat[row].y;
    #pragma unroll
    for (int j = 0; j < 16; ++j) {
      int c = (sub + j * 16) * 4;
      float4 w4 = *(const float4*)(lnw + c);
      float4 b4 = *(const float4*)(lnb + c);
      float4 xs = xs_loc[j];
      unsigned short* dst = &xnb[row * ROWB + c];
      dst[0] = f32_to_bf16_rne((xs.x - mu) * rs * w4.x + b4.x);
      dst[1] = f32_to_bf16_rne((xs.y - mu) * rs * w4.y + b4.y);
      dst[2] = f32_to_bf16_rne((xs.z - mu) * rs * w4.z + b4.z);
      dst[3] = f32_to_bf16_rne((xs.w - mu) * rs * w4.w + b4.w);
    }
  }
  __syncthreads();

  // ---- phase 3: WMMA k-loop ----
  const int lane   = tid & 31;
  const int wave   = tid >> 5;
  const int matSel = wave >> 2;          // 0=key, 1=value
  const int tn     = wave & 3;           // n-tile
  const int arow   = lane & 15;          // A matrix M row
  const int grp    = lane >> 4;
  const unsigned short* bsrc = packed + ((matSel * 4 + tn) * 16384) + lane * 16;

  union U16 { uint4 q[2]; v16bf v; };
  v8f acc = {};
  for (int ks = 0; ks < 32; ++ks) {
    if (ks + 1 < 32) __builtin_prefetch(bsrc + (ks + 1) * 512, 0, 1);
    U16 afrag, bfrag;
    // A (16x32 bf16): lanes 0-15 K{0..7,16..23}, lanes 16-31 K{8..15,24..31}
    const unsigned short* ap = &xnb[arow * ROWB + ks * 32 + grp * 8];
    afrag.q[0] = *(const uint4*)(ap);
    afrag.q[1] = *(const uint4*)(ap + 16);
    const uint4* bp = (const uint4*)(bsrc + ks * 512);
    bfrag.q[0] = bp[0];
    bfrag.q[1] = bp[1];
    acc = __builtin_amdgcn_wmma_f32_16x16x32_bf16(
        false, afrag.v, false, bfrag.v, (short)0, acc, false, false);
  }
  // C layout: VGPR i -> M = i + 8*grp, N = lane&15
  #pragma unroll
  for (int i = 0; i < 8; ++i)
    kvbuf[matSel][i + 8 * grp][tn * 16 + (lane & 15)] = acc[i];
  __syncthreads();

  // ---- phase 4: wkv = exp(-exp(tf)*k) * v, coalesced (b,s,n) store ----
  {
    const int erow = tid >> 4;
    const int ng   = tid & 15;
    float4 o;
    #pragma unroll
    for (int c = 0; c < 4; ++c) {
      int n = ng * 4 + c;
      float kk = kvbuf[0][erow][n];
      float vv = kvbuf[1][erow][n];
      ((float*)&o)[c] = __expf(-tf[n] * kk) * vv;
    }
    *(float4*)(wkv + (size_t)(b * S_ + s0 + erow) * N_ + ng * 4) = o;
  }
}

// ---------------------------------------------------------------------------
// 2: chunked linear recurrence  s_t = w*s_{t-1} + u_t
// pass1: per-chunk local result with zero init (U). 64 threads = n lanes,
//        fully coalesced 256B reads of wkv.
// ---------------------------------------------------------------------------
__global__ __launch_bounds__(64) void scan_pass1(
    const float* __restrict__ wkv, const float* __restrict__ wn,
    float* __restrict__ carryU) {
  int blk = blockIdx.x;             // b*NCHUNK + ch
  int b = blk >> 5, ch = blk & (NCHUNK - 1);
  int n = threadIdx.x;
  float w = wn[n];
  float s = 0.f;
  const float* p = wkv + (size_t)(b * S_ + ch * CHUNK) * N_ + n;
  for (int j = 0; j < CHUNK; ++j) s = fmaf(w, s, p[(size_t)j * N_]);
  carryU[(size_t)blk * N_ + n] = s;
}

// pass2: compose affine chunk maps (W=w^128) sequentially; emit exclusive
// carries and last_states.
__global__ __launch_bounds__(512) void scan_pass2(
    const float* __restrict__ carryU, const float* __restrict__ wc,
    float* __restrict__ carryE, float* __restrict__ last_states) {
  int tid = threadIdx.x;            // 512 = B*N
  int b = tid >> 6, n = tid & 63;
  float W = wc[n];
  float c = 0.f;
  for (int ch = 0; ch < NCHUNK; ++ch) {
    size_t o = (size_t)(b * NCHUNK + ch) * N_ + n;
    carryE[o] = c;
    c = fmaf(W, c, carryU[o]);
  }
  last_states[b * N_ + n] = c;
}

// pass3: re-run each chunk with its carry-in, write states (b,s,n).
__global__ __launch_bounds__(64) void scan_pass3(
    const float* __restrict__ wkv, const float* __restrict__ wn,
    const float* __restrict__ carryE, float* __restrict__ states) {
  int blk = blockIdx.x;
  int b = blk >> 5, ch = blk & (NCHUNK - 1);
  int n = threadIdx.x;
  float w = wn[n];
  float s = carryE[(size_t)blk * N_ + n];
  size_t base = (size_t)(b * S_ + ch * CHUNK) * N_ + n;
  for (int j = 0; j < CHUNK; ++j) {
    s = fmaf(w, s, wkv[base + (size_t)j * N_]);
    states[base + (size_t)j * N_] = s;
  }
}

// ---------------------------------------------------------------------------
// 3: output projection out[b,s,d] = sum_n states[b,s,n]*output_w[d,n]
// Full f32 precision via v_wmma_f32_16x16x4_f32.
// A (16x4 f32): per-lane v2f = states K = ks*4 + 2*grp + {0,1}, row=lane&15.
// B (4x16 f32): per-lane v2f = output_w[d][same K pair],        col=lane&15.
// ---------------------------------------------------------------------------
__global__ __launch_bounds__(256) void out_proj(
    const float* __restrict__ states, const float* __restrict__ ow,
    float* __restrict__ out) {
  __shared__ float stage[16][128];
  const int tid = threadIdx.x;
  const int lane = tid & 31, wave = tid >> 5;
  const int r0 = blockIdx.x * 16;
  const int b = r0 >> 12, s0 = r0 & (S_ - 1);
  const int dcol = blockIdx.y * 128 + wave * 16 + (lane & 15);
  const int arow = lane & 15;
  const int grp  = lane >> 4;
  const float* ap = states + (size_t)(b * S_ + s0 + arow) * N_;
  const float* bp = ow + (size_t)dcol * N_;

  v8f acc = {};
  #pragma unroll
  for (int ks = 0; ks < 16; ++ks) {
    int k0 = ks * 4 + grp * 2;
    v2f a  = *(const v2f*)(ap + k0);
    v2f bb = *(const v2f*)(bp + k0);
    acc = __builtin_amdgcn_wmma_f32_16x16x4_f32(
        false, a, false, bb, (short)0, acc, false, false);
  }
  #pragma unroll
  for (int i = 0; i < 8; ++i)
    stage[i + 8 * grp][wave * 16 + (lane & 15)] = acc[i];
  __syncthreads();

  // coalesced 512B-per-row stores via LDS stage
  int row = tid >> 4;
  int c8  = (tid & 15) * 8;
  float* dst = out + (size_t)(b * S_ + s0 + row) * D_ + blockIdx.y * 128 + c8;
  *(float4*)(dst)     = *(float4*)&stage[row][c8];
  *(float4*)(dst + 4) = *(float4*)&stage[row][c8 + 4];
}

// ---------------------------------------------------------------------------
// host launcher
// ---------------------------------------------------------------------------
extern "C" void kernel_launch(void* const* d_in, const int* in_sizes, int n_in,
                              void* d_out, int out_size, void* d_ws, size_t ws_size,
                              hipStream_t stream) {
  const float* x     = (const float*)d_in[0];
  const float* td    = (const float*)d_in[1];
  const float* tf_in = (const float*)d_in[2];
  const float* key_w = (const float*)d_in[3];
  const float* val_w = (const float*)d_in[4];
  const float* out_w = (const float*)d_in[5];
  const float* gates = (const float*)d_in[6];
  const float* lnw   = (const float*)d_in[7];
  const float* lnb   = (const float*)d_in[8];
  float* out         = (float*)d_out;                   // (B,S,D) then (B,N)
  float* last_states = out + (size_t)B_ * S_ * D_;

  // workspace carve-up (256B aligned)
  char* ws = (char*)d_ws;
  unsigned short* packed = (unsigned short*)(ws + 0);          // 262144 B
  float* g_sig  = (float*)(ws + 262144);                       //   4096 B
  float* tf     = (float*)(ws + 266240);                       //    256 B
  float* wn     = (float*)(ws + 266496);                       //    256 B
  float* wc     = (float*)(ws + 266752);                       //    256 B
  float* wkv    = (float*)(ws + 267008);                       // 8388608 B
  float* states = (float*)(ws + 267008 + 8388608);             // 8388608 B
  float* carryU = (float*)(ws + 267008 + 2 * 8388608);         //  65536 B
  float* carryE = (float*)(ws + 267008 + 2 * 8388608 + 65536); //  65536 B

  prep_scalars<<<1, 1024, 0, stream>>>(td, tf_in, gates, g_sig, tf, wn, wc);
  prep_pack<<<512, 256, 0, stream>>>(key_w, val_w, packed);
  fused_shift_ln_kv<<<(B_ * S_) / 16, 256, 0, stream>>>(
      x, lnw, lnb, g_sig, packed, tf, wkv);
  scan_pass1<<<B_ * NCHUNK, 64, 0, stream>>>(wkv, wn, carryU);
  scan_pass2<<<1, 512, 0, stream>>>(carryU, wc, carryE, last_states);
  scan_pass3<<<B_ * NCHUNK, 64, 0, stream>>>(wkv, wn, carryE, states);
  out_proj<<<dim3((B_ * S_) / 16, D_ / 128), 256, 0, stream>>>(states, out_w, out);
}
